// LocalAttention_17334488006719
// MI455X (gfx1250) — compile-verified, matrix operand order checked by reading, not run on hardware
//
#include <hip/hip_runtime.h>
#include <hip/hip_bf16.h>
#include <stdint.h>

// Problem constants (from reference): x [4, 2048, 1024]
constexpr int B_SZ = 4;
constexpr int T_SZ = 2048;
constexpr int C_SZ = 1024;
constexpr int BT   = B_SZ * T_SZ;      // 8192

typedef __attribute__((ext_vector_type(16))) __bf16 bf16x16;
typedef __attribute__((ext_vector_type(8)))  float  f32x8;

__device__ __forceinline__ uint16_t f2bf(float f) {
    uint32_t u = __float_as_uint(f);
    u += 0x7FFFu + ((u >> 16) & 1u);   // round-to-nearest-even
    return (uint16_t)(u >> 16);
}

// ---- CDNA5 async LDS copy path (ASYNCcnt, CDNA5 ISA) ----
__device__ __forceinline__ void async_tile_b128(const uint16_t* g, const uint16_t* l) {
    asm volatile("global_load_async_to_lds_b128 %0, %1, off"
                 :: "v"((uint32_t)(uintptr_t)l), "v"(g)
                 : "memory");
}
__device__ __forceinline__ void wait_async0() { asm volatile("s_wait_asynccnt 0x0" ::: "memory"); }
__device__ __forceinline__ void wait_async2() { asm volatile("s_wait_asynccnt 0x2" ::: "memory"); }

// 16x32 bf16 WMMA fragment (A layout; also valid for B stored as [N, K]).
// Lanes 0-15 hold K={0..7,16..23}, lanes 16-31 hold K={8..15,24..31}:
// two b128 loads per lane.
__device__ __forceinline__ bf16x16 load_frag(const uint16_t* __restrict__ base,
                                             int ld, int kk) {
    const int lane = threadIdx.x & 31;
    const int r    = lane & 15;
    const int kb   = (lane >> 4) ? 8 : 0;
    const uint16_t* p = base + (size_t)r * ld + kk + kb;
    union { uint4 u[2]; bf16x16 v; } x;
    x.u[0] = *(const uint4*)(p);
    x.u[1] = *(const uint4*)(p + 16);
    return x.v;
}

__global__ void cast_f32_to_bf16(const float* __restrict__ src,
                                 uint16_t* __restrict__ dst, int n) {
    int i = blockIdx.x * blockDim.x + threadIdx.x;
    const int stride = gridDim.x * blockDim.x;
    for (; i < n; i += stride) dst[i] = f2bf(src[i]);
}

// C[m,n] = sum_k A[m,k] * Bw[n,k] + bias[n]   (M=8192, N=K=1024, all constexpr)
// Block tile 128x128 (8 waves: 4 in M x 2 in N), wave tile 32x64 -> each B
// fragment feeds two WMMAs.  B tiles staged via async-LDS, 4 buffers, fully
// branchless k-loop (wrapped prefetch index), one barrier per k-step.
// MODE 0: bf16 [M,N]; 1: bf16 transposed-per-batch [b, N, T] (V^T); 2: f32.
constexpr int BPAD = 40;   // 80-byte LDS row stride -> conflict-free b128 gathers

template<int MODE>
__global__ void __launch_bounds__(256)
wmma_gemm_bias(const uint16_t* __restrict__ A, const uint16_t* __restrict__ Bw,
               const float* __restrict__ bias, void* __restrict__ out)
{
    constexpr int M = BT, N = C_SZ, K = C_SZ;
    constexpr int steps = K >> 5;                 // 32
    __shared__ uint16_t sB[4][128 * BPAD];        // 4 x 10 KB

    const int wave  = threadIdx.x >> 5;
    const int lane  = threadIdx.x & 31;
    const int mw    = wave >> 1;                  // 0..3
    const int nw    = wave & 1;                   // 0..1
    const int mRow0 = blockIdx.y * 128 + mw * 32;
    const int nColB = blockIdx.x * 128;

    // Staging map: 512 x 16B chunks per k-step, 2 per thread.
    const int n0 = threadIdx.x >> 2;              // 0..63
    const int k0 = (threadIdx.x & 3) * 8;
    const int n1 = n0 + 64;                       // 64..127
    const uint16_t* gB0 = Bw + (size_t)(nColB + n0) * K + k0;
    const uint16_t* gB1 = Bw + (size_t)(nColB + n1) * K + k0;
    const int l0 = n0 * BPAD + k0;
    const int l1 = n1 * BPAD + k0;

    async_tile_b128(gB0, &sB[0][l0]);
    async_tile_b128(gB1, &sB[0][l1]);

    const uint16_t* gA0 = A + (size_t)mRow0 * K;
    const uint16_t* gA1 = gA0 + (size_t)16 * K;
    bf16x16 aN0 = load_frag(gA0, K, 0);
    bf16x16 aN1 = load_frag(gA1, K, 0);

    f32x8 acc0[4] = {}, acc1[4] = {};
    #pragma unroll 4
    for (int i = 0; i < steps; ++i) {
        // Branchless prefetch: wraps on the final steps; the wrapped tile's
        // buffer is two barriers away from its last reader -> race-free.
        const int nx = ((i + 1) & (steps - 1)) * 32;
        const int nb = (i + 1) & 3;
        async_tile_b128(gB0 + nx, &sB[nb][l0]);
        async_tile_b128(gB1 + nx, &sB[nb][l1]);
        wait_async2();               // tile i resident; tile i+1 in flight
        __syncthreads();

        bf16x16 a0 = aN0, a1 = aN1;
        aN0 = load_frag(gA0, K, nx);               // overlap with WMMAs
        aN1 = load_frag(gA1, K, nx);

        const uint16_t* lb = &sB[i & 3][nw * 64 * BPAD];
        #pragma unroll
        for (int j = 0; j < 4; ++j) {
            bf16x16 b = load_frag(lb + j * 16 * BPAD, BPAD, 0);
            acc0[j] = __builtin_amdgcn_wmma_f32_16x16x32_bf16(
                false, a0, false, b, (short)0, acc0[j], false, false);
            acc1[j] = __builtin_amdgcn_wmma_f32_16x16x32_bf16(
                false, a1, false, b, (short)0, acc1[j], false, false);
        }
    }

    const int n    = lane & 15;
    const int roff = (lane >> 4) ? 8 : 0;
    #pragma unroll
    for (int half = 0; half < 2; ++half) {
        const f32x8* acc = half ? acc1 : acc0;
        const int rowBase = mRow0 + half * 16 + roff;
        #pragma unroll
        for (int j = 0; j < 4; ++j) {
            const int col = nColB + nw * 64 + j * 16 + n;
            const float bv = bias[col];
            #pragma unroll
            for (int r = 0; r < 8; ++r) {
                const int row = rowBase + r;
                const float v = acc[j][r] + bv;
                if constexpr (MODE == 0) {
                    ((uint16_t*)out)[(size_t)row * N + col] = f2bf(v);
                } else if constexpr (MODE == 1) {  // store V transposed
                    const int bb = row >> 11;
                    const int s  = row & 2047;
                    ((uint16_t*)out)[((size_t)bb * N + col) * T_SZ + s] = f2bf(v);
                } else {
                    ((float*)out)[(size_t)row * N + col] = v;
                }
            }
        }
    }
}

// Flash attention: one block = 16 query rows of one batch. 8 waves.
// Q tile (16x1024, 32 KB) staged ONCE per block into LDS via async copies.
// Waves 0-3 compute S tiles into LDS; all waves do shuffle-based online
// softmax; all 8 waves own 128 channels of the 16x1024 ctx accumulator.
constexpr int QPAD = 1032;  // 2064-byte row stride -> distinct bank groups

__global__ void __launch_bounds__(256)
flash_attn(const uint16_t* __restrict__ Q, const uint16_t* __restrict__ Kmat,
           const uint16_t* __restrict__ Vt, uint16_t* __restrict__ Ctx)
{
    __shared__ uint16_t sQ[16][QPAD];  // ~33 KB, reused for all 32 key blocks
    __shared__ float    sS[16][64];    // score block (f32)
    __shared__ uint16_t sP[16][64];    // exp(S - m) as bf16
    __shared__ float    sM[16], sL[16], sA[16];

    const int wave  = threadIdx.x >> 5;
    const int lane  = threadIdx.x & 31;
    const int bIdx  = blockIdx.x >> 7;          // / (T/16)
    const int qt    = blockIdx.x & 127;
    const int qRow0 = bIdx * T_SZ + qt * 16;

    // ---- Stage Q tile: 2048 x 16B chunks, 8 per thread, ASYNCcnt path ----
    #pragma unroll
    for (int t = 0; t < 8; ++t) {
        const int m   = threadIdx.x + t * 256;  // 0..2047
        const int row = m >> 7;
        const int col = (m & 127) * 8;
        async_tile_b128(Q + (size_t)(qRow0 + row) * C_SZ + col, &sQ[row][col]);
    }
    if (threadIdx.x < 16) { sM[threadIdx.x] = -3.0e38f; sL[threadIdx.x] = 0.0f; }
    wait_async0();
    __syncthreads();

    f32x8 acc[8] = {};                 // 16 rows x 128 channels per wave
    const float scale = 0.03125f;      // 1/sqrt(1024)
    constexpr int NKB = T_SZ / 64;

    for (int kb = 0; kb < NKB; ++kb) {
        // ---- Phase A: S = Q * K^T for this 64-key block (waves 0..3) ----
        if (wave < 4) {
            const int key0 = bIdx * T_SZ + kb * 64 + wave * 16;
            const uint16_t* Kbase = Kmat + (size_t)key0 * C_SZ;
            if (kb + 1 < NKB)
                __builtin_prefetch(Kbase + (size_t)64 * C_SZ, 0, 1);
            f32x8 s = {};
            bf16x16 bN = load_frag(Kbase, C_SZ, 0);
            for (int kk = 0; kk < C_SZ; kk += 32) {
                bf16x16 a = load_frag(&sQ[0][0], QPAD, kk);   // ds path
                bf16x16 b = bN;
                if (kk + 32 < C_SZ) bN = load_frag(Kbase, C_SZ, kk + 32);
                s = __builtin_amdgcn_wmma_f32_16x16x32_bf16(
                    false, a, false, b, (short)0, s, false, false);
            }
            const int n    = lane & 15;
            const int roff = (lane >> 4) ? 8 : 0;
            #pragma unroll
            for (int r = 0; r < 8; ++r)
                sS[roff + r][wave * 16 + n] = s[r] * scale;
        }
        __syncthreads();

        // ---- Phase B: online softmax; wave w handles rows 2w, 2w+1 ----
        {
            const int row = wave * 2 + (lane >> 4);
            const int c0  = lane & 15;
            float v0 = sS[row][c0],      v1 = sS[row][c0 + 16];
            float v2 = sS[row][c0 + 32], v3 = sS[row][c0 + 48];
            float mx = fmaxf(fmaxf(v0, v1), fmaxf(v2, v3));
            #pragma unroll
            for (int d = 1; d < 16; d <<= 1) mx = fmaxf(mx, __shfl_xor(mx, d));
            const float mold  = sM[row];
            const float mnew  = fmaxf(mold, mx);
            const float alpha = __expf(mold - mnew);
            const float p0 = __expf(v0 - mnew), p1 = __expf(v1 - mnew);
            const float p2 = __expf(v2 - mnew), p3 = __expf(v3 - mnew);
            sP[row][c0]      = f2bf(p0);  sP[row][c0 + 16] = f2bf(p1);
            sP[row][c0 + 32] = f2bf(p2);  sP[row][c0 + 48] = f2bf(p3);
            float sum = p0 + p1 + p2 + p3;
            #pragma unroll
            for (int d = 1; d < 16; d <<= 1) sum += __shfl_xor(sum, d);
            if (c0 == 0) {
                sM[row] = mnew;
                sL[row] = alpha * sL[row] + sum;
                sA[row] = alpha;
            }
        }
        __syncthreads();

        // ---- Phase C: rescale ctx and accumulate P @ V^T (all 8 waves) ----
        {
            const int roff = (lane >> 4) ? 8 : 0;
            float al[8];
            #pragma unroll
            for (int r = 0; r < 8; ++r) al[r] = sA[roff + r];
            #pragma unroll
            for (int j = 0; j < 8; ++j) {
                #pragma unroll
                for (int r = 0; r < 8; ++r) acc[j][r] *= al[r];
            }
            const uint16_t* Vbase =
                Vt + ((size_t)bIdx * C_SZ + wave * 128) * T_SZ + kb * 64;
            bf16x16 a0 = load_frag(&sP[0][0], 64, 0);
            bf16x16 a1 = load_frag(&sP[0][0], 64, 32);

            // Depth-4 rolling software pipeline over the 16 (kks, j)
            // fragments: only 4 V fragments live at once, each WMMA waits on
            // a load issued 4 WMMAs earlier -> steady overlap at bounded
            // register pressure (instead of 8-deep then fully serialized).
            auto fragV = [&](int idx) {
                const int kks = idx >> 3;          // 0..1
                const int j   = idx & 7;           // 0..7
                return load_frag(Vbase + (size_t)j * 16 * T_SZ, T_SZ, kks * 32);
            };
            bf16x16 vb[4];
            #pragma unroll
            for (int p = 0; p < 4; ++p) vb[p] = fragV(p);
            #pragma unroll
            for (int idx = 0; idx < 16; ++idx) {
                bf16x16 b = vb[idx & 3];
                if (idx + 4 < 16) vb[idx & 3] = fragV(idx + 4);
                acc[idx & 7] = __builtin_amdgcn_wmma_f32_16x16x32_bf16(
                    false, (idx < 8) ? a0 : a1, false, b, (short)0,
                    acc[idx & 7], false, false);
            }
        }
    }

    // ---- Final: ctx / l, store bf16 ----
    const int n    = lane & 15;
    const int roff = (lane >> 4) ? 8 : 0;
    float li[8];
    #pragma unroll
    for (int r = 0; r < 8; ++r) li[r] = 1.0f / sL[roff + r];
    #pragma unroll
    for (int j = 0; j < 8; ++j) {
        const int ch = wave * 128 + j * 16 + n;
        #pragma unroll
        for (int r = 0; r < 8; ++r) {
            const int row = qRow0 + roff + r;
            Ctx[(size_t)row * C_SZ + ch] = f2bf(acc[j][r] * li[r]);
        }
    }
}

extern "C" void kernel_launch(void* const* d_in, const int* in_sizes, int n_in,
                              void* d_out, int out_size, void* d_ws, size_t ws_size,
                              hipStream_t stream) {
    const float* x  = (const float*)d_in[0];
    const float* Wq = (const float*)d_in[1];
    const float* bq = (const float*)d_in[2];
    const float* Wk = (const float*)d_in[3];
    const float* bk = (const float*)d_in[4];
    const float* Wv = (const float*)d_in[5];
    const float* bv = (const float*)d_in[6];
    const float* Wo = (const float*)d_in[7];
    const float* bo = (const float*)d_in[8];

    // Workspace layout (bf16, ~88 MB total)
    uint8_t* ws = (uint8_t*)d_ws;
    uint16_t* xb   = (uint16_t*)ws;                            // 8192x1024
    uint16_t* wqb  = (uint16_t*)(ws + (size_t)16 * 1024 * 1024);
    uint16_t* wkb  = wqb + (size_t)C_SZ * C_SZ;
    uint16_t* wvb  = wkb + (size_t)C_SZ * C_SZ;
    uint16_t* wob  = wvb + (size_t)C_SZ * C_SZ;
    uint16_t* qb   = wob + (size_t)C_SZ * C_SZ;                // [BT, C]
    uint16_t* kbuf = qb  + (size_t)BT * C_SZ;                  // [BT, C]
    uint16_t* vtb  = kbuf + (size_t)BT * C_SZ;                 // [B, C, T]
    uint16_t* ctxb = vtb + (size_t)BT * C_SZ;                  // [BT, C]

    cast_f32_to_bf16<<<1024, 256, 0, stream>>>(x,  xb,  BT * C_SZ);
    cast_f32_to_bf16<<<256,  256, 0, stream>>>(Wq, wqb, C_SZ * C_SZ);
    cast_f32_to_bf16<<<256,  256, 0, stream>>>(Wk, wkb, C_SZ * C_SZ);
    cast_f32_to_bf16<<<256,  256, 0, stream>>>(Wv, wvb, C_SZ * C_SZ);
    cast_f32_to_bf16<<<256,  256, 0, stream>>>(Wo, wob, C_SZ * C_SZ);

    dim3 gg(C_SZ / 128, BT / 128);   // (8, 64)
    wmma_gemm_bias<0><<<gg, 256, 0, stream>>>(xb, wqb, bq, qb);
    wmma_gemm_bias<0><<<gg, 256, 0, stream>>>(xb, wkb, bk, kbuf);
    wmma_gemm_bias<1><<<gg, 256, 0, stream>>>(xb, wvb, bv, vtb);

    flash_attn<<<BT / 16, 256, 0, stream>>>(qb, kbuf, vtb, ctxb);

    wmma_gemm_bias<2><<<gg, 256, 0, stream>>>(ctxb, wob, bo, d_out);
}
